// ClassDependentKeyAttention_26276609917692
// MI455X (gfx1250) — matrix operand order repaired and checked by reading, not compiled, not run on hardware
//
#include <hip/hip_runtime.h>
#include <hip/hip_bf16.h>

#define BB 8
#define SS 2048
#define CC 32
#define NH 4
#define HD 64
#define MTOT (BB*SS)   // 16384 rows

typedef __attribute__((ext_vector_type(16))) __bf16 v16bf;
typedef __attribute__((ext_vector_type(8)))  float  v8f;
typedef __attribute__((ext_vector_type(4)))  float  f32x4;
typedef __attribute__((ext_vector_type(8)))  unsigned short us8;
typedef __attribute__((ext_vector_type(4)))  unsigned short us4;
typedef __attribute__((ext_vector_type(4)))  unsigned int   u32x4;
typedef __attribute__((ext_vector_type(8)))  int            i32x8;
typedef __attribute__((ext_vector_type(4)))  int            i32x4;

#if __has_builtin(__builtin_amdgcn_global_load_async_to_lds_b128)
#define HAVE_ASYNC_LDS 1
typedef __attribute__((address_space(1))) i32x4 as1_i32x4;
typedef __attribute__((address_space(3))) i32x4 as3_i32x4;
#endif
#if __has_builtin(__builtin_amdgcn_tensor_load_to_lds)
#define HAVE_TDM 1
#endif

__device__ __forceinline__ unsigned short f2bf(float f) {
  union { float f; unsigned u; } x; x.f = f;
  unsigned r = x.u + 0x7FFFu + ((x.u >> 16) & 1u);
  return (unsigned short)(r >> 16);
}

__device__ __forceinline__ v16bf frag16(const unsigned short* p0, const unsigned short* p1) {
  union { v16bf v; us8 h[2]; } u;
  u.h[0] = *(const us8*)p0;
  u.h[1] = *(const us8*)p1;
  return u.v;
}

// 16-byte global -> LDS copy: async DMA path if available, else through VGPRs.
__device__ __forceinline__ void cp16(unsigned short* l, const unsigned short* g) {
#ifdef HAVE_ASYNC_LDS
  __builtin_amdgcn_global_load_async_to_lds_b128((as1_i32x4*)g, (as3_i32x4*)l, 0, 0);
#else
  *(us8*)l = *(const us8*)g;
#endif
}
__device__ __forceinline__ void cp_wait() {
#ifdef HAVE_ASYNC_LDS
#if __has_builtin(__builtin_amdgcn_s_wait_asynccnt)
  __builtin_amdgcn_s_wait_asynccnt(0);
#else
  asm volatile("s_wait_asynccnt 0x0" ::: "memory");
#endif
#endif
}

#ifdef HAVE_TDM
// 2D tile load via Tensor Data Mover: tile_y rows x tile_x bf16 elems, row stride
// stride_e elems in memory; LDS rows padded +8 ushorts (pad 4 DW every 16 DW).
__device__ __forceinline__ void tdm_load_2d(unsigned lds_off, const void* gaddr,
                                            int tile_x, int tile_y, int stride_e) {
  unsigned long long ga = (unsigned long long)gaddr;
  u32x4 g0;
  g0.x = 1u;                                           // count=1, user mode
  g0.y = lds_off;                                      // lds_addr
  g0.z = (unsigned)(ga & 0xFFFFFFFFu);                 // global_addr[31:0]
  g0.w = (unsigned)((ga >> 32) & 0x01FFFFFFu) | 0x80000000u;  // addr[56:32] | type=2
  int td0 = stride_e;          // tensor_dim0 (elems per row; tile in-bounds)
  int td1 = 1 << 20;           // tensor_dim1 (large; tile in-bounds)
  i32x8 g1;
  g1[0] = (1 << 16) | (1 << 20) | (3 << 22) | (3 << 25); // 2B elems, pad_en, 16DW interval, +4DW
  g1[1] = (td0 & 0xFFFF) << 16;                          // abar=0 | tensor_dim0 lo
  g1[2] = ((td0 >> 16) & 0xFFFF) | ((td1 & 0xFFFF) << 16);
  g1[3] = ((td1 >> 16) & 0xFFFF) | ((tile_x & 0xFFFF) << 16);   // | tile_dim0
  g1[4] = (tile_y & 0xFFFF);                             // tile_dim1 | tile_dim2=0
  g1[5] = stride_e;                                      // tensor_dim0_stride lo32
  g1[6] = 0;
  g1[7] = 0;
  i32x4 z4 = {0, 0, 0, 0};
  i32x8 z8 = {0, 0, 0, 0, 0, 0, 0, 0};
  __builtin_amdgcn_tensor_load_to_lds(g0, g1, z4, z4, z8, 0);
}
#endif

// ---------------------------------------------------------------- convert
__global__ void k_convert(const float* __restrict__ src,
                          unsigned short* __restrict__ dst, int n4) {
  int i = blockIdx.x * blockDim.x + threadIdx.x;
  if (i < n4) {
    f32x4 v = ((const f32x4*)src)[i];
    us4 o; o.x = f2bf(v.x); o.y = f2bf(v.y); o.z = f2bf(v.z); o.w = f2bf(v.w);
    ((us4*)dst)[i] = o;
  }
}

// ---------------------------------------------------------------- GEMM: out = A @ W^T + bias
// double-buffered LDS, 1 barrier per K-step. block=256 (8 waves), tile 64x128.
template<bool F32OUT>
__global__ void k_gemm(const unsigned short* __restrict__ A,
                       const unsigned short* __restrict__ W,
                       const float* __restrict__ bias,
                       void* __restrict__ out) {
  __shared__ unsigned short As[2][64 * 40];
  __shared__ unsigned short Bs[2][128 * 40];
  const int t = threadIdx.x, lane = t & 31, wv = t >> 5;
  const int wm = wv & 3, wn = wv >> 2;
  const int rowBase = blockIdx.x * 64, colBase = blockIdx.y * 128;
  const int ln = lane & 15, hf = lane >> 4;

  auto stage = [&](int buf, int kk) {
    { int row = t >> 2, cg = (t & 3) * 8;
      cp16(&As[buf][row*40 + cg], &A[(rowBase+row)*256 + kk + cg]); }
    #pragma unroll
    for (int i = 0; i < 2; ++i) {
      int lin = i*2048 + t*8, row = lin >> 5, cg = lin & 31;
      cp16(&Bs[buf][row*40 + cg], &W[(colBase+row)*256 + kk + cg]);
    }
  };

  v8f acc[4] = {};
  stage(0, 0);
  cp_wait();
  __syncthreads();
  for (int st = 0; st < 8; ++st) {
    const int cur = st & 1;
    if (st < 7) stage(1 - cur, (st + 1) * 32);
    const unsigned short* ap = &As[cur][(wm*16 + ln)*40 + hf*8];
    v16bf a = frag16(ap, ap + 16);
    v16bf bf[4];
    #pragma unroll
    for (int j = 0; j < 4; ++j) {
      const unsigned short* bp = &Bs[cur][(wn*64 + j*16 + ln)*40 + hf*8];
      bf[j] = frag16(bp, bp + 16);
    }
    #pragma unroll
    for (int j = 0; j < 4; ++j)
      acc[j] = __builtin_amdgcn_wmma_f32_16x16x32_bf16(false, a, false, bf[j],
                                                       (short)0, acc[j], false, false);
    cp_wait();
    __syncthreads();
  }
  #pragma unroll
  for (int j = 0; j < 4; ++j) {
    int col = colBase + wn*64 + j*16 + ln;
    float bv = bias[col];
    #pragma unroll
    for (int r = 0; r < 8; ++r) {
      int row = rowBase + wm*16 + r + hf*8;
      float val = acc[j][r] + bv;
      if (F32OUT) ((float*)out)[row*256 + col] = val;
      else        ((unsigned short*)out)[row*256 + col] = f2bf(val);
    }
  }
}

// ---------------------------------------------------------------- fused class keys
// k[row,:] = sum_c cond[row,c] * (x[row,:] @ Wk[c]^T + Wkb[c,:])
// x-tile (64x256) resident in LDS; Wk tiles double-buffered (TDM when available).
__global__ void k_classkey(const unsigned short* __restrict__ X,
                           const unsigned short* __restrict__ Wk,   // 32 x 256 x 256 bf16
                           const float* __restrict__ Wkb,           // 32 x 256 f32
                           const float* __restrict__ cond,          // MTOT x 32 f32
                           unsigned short* __restrict__ Kout) {
  __shared__ unsigned short As[64 * 264];
  __shared__ unsigned short Bs[2][128 * 40];
  __shared__ float Cs[64 * 32];
  const int t = threadIdx.x, lane = t & 31, wv = t >> 5;
  const int wm = wv & 3, wn = wv >> 2;
  const int rowBase = blockIdx.x * 64, colBase = blockIdx.y * 128;
  const int ln = lane & 15, hf = lane >> 4;

  auto stageB = [&](int buf, int s) {       // s = c*8 + kstep
    const int c = s >> 3, kk = (s & 7) * 32;
#ifdef HAVE_TDM
    if (wv == 0) {
      unsigned lds_off = (unsigned)(size_t)&Bs[buf][0];
      tdm_load_2d(lds_off, &Wk[(size_t)(c*256 + colBase)*256 + kk], 32, 128, 256);
    }
#else
    #pragma unroll
    for (int i = 0; i < 2; ++i) {
      int lin = i*2048 + t*8, row = lin >> 5, cg = lin & 31;
      cp16(&Bs[buf][row*40 + cg], &Wk[(size_t)(c*256 + colBase + row)*256 + kk + cg]);
    }
#endif
  };
  auto stage_wait = [&]() {
#ifdef HAVE_TDM
    if (wv == 0) __builtin_amdgcn_s_wait_tensorcnt(0);
#else
    cp_wait();
#endif
  };

  #pragma unroll
  for (int i = 0; i < 8; ++i) {
    int lin = i*2048 + t*8, row = lin >> 8, col = lin & 255;
    cp16(&As[row*264 + col], &X[(rowBase+row)*256 + col]);
  }
  { int row = t >> 2, cg = (t & 3) * 8;
    *(f32x4*)&Cs[row*32 + cg]     = *(const f32x4*)&cond[(rowBase+row)*32 + cg];
    *(f32x4*)&Cs[row*32 + cg + 4] = *(const f32x4*)&cond[(rowBase+row)*32 + cg + 4]; }

  stageB(0, 0);
  cp_wait();
  stage_wait();
  __syncthreads();

  v8f kacc[4] = {};
  v8f cacc[4] = {};
  for (int s = 0; s < 256; ++s) {           // 32 classes x 8 K-steps
    const int cur = s & 1, kk = (s & 7) * 32;
    if (s < 255) stageB(1 - cur, s + 1);
    const unsigned short* ap = &As[(wm*16 + ln)*264 + kk + hf*8];
    v16bf a = frag16(ap, ap + 16);
    v16bf bf[4];
    #pragma unroll
    for (int j = 0; j < 4; ++j) {
      const unsigned short* bp = &Bs[cur][(wn*64 + j*16 + ln)*40 + hf*8];
      bf[j] = frag16(bp, bp + 16);
    }
    #pragma unroll
    for (int j = 0; j < 4; ++j)
      cacc[j] = __builtin_amdgcn_wmma_f32_16x16x32_bf16(false, a, false, bf[j],
                                                        (short)0, cacc[j], false, false);
    if ((s & 7) == 7) {                     // end of class: cond-scaled accumulate
      const int c = s >> 3;
      #pragma unroll
      for (int j = 0; j < 4; ++j) {
        int col = colBase + wn*64 + j*16 + ln;
        float kb = Wkb[c*256 + col];
        #pragma unroll
        for (int r = 0; r < 8; ++r) {
          int m = wm*16 + r + hf*8;
          kacc[j][r] += Cs[m*32 + c] * (cacc[j][r] + kb);
          cacc[j][r] = 0.f;
        }
      }
    }
    stage_wait();
    __syncthreads();
  }
  #pragma unroll
  for (int j = 0; j < 4; ++j) {
    int col = colBase + wn*64 + j*16 + ln;
    #pragma unroll
    for (int r = 0; r < 8; ++r) {
      int row = rowBase + wm*16 + r + hf*8;
      Kout[row*256 + col] = f2bf(kacc[j][r]);
    }
  }
}

// ---------------------------------------------------------------- flash attention
// block=128 (4 waves), 16 q-rows/wave, key tiles of 32, double-buffered K/V staging.
__global__ void k_attn(const unsigned short* __restrict__ Q,
                       const unsigned short* __restrict__ K,
                       const unsigned short* __restrict__ V,
                       unsigned short* __restrict__ O) {
  __shared__ unsigned short Kt[2][32 * 72];   // [key][d]  B-layout for Q@K^T
  __shared__ unsigned short Vt[2][64 * 40];   // [d][key]  B-layout for P@V
  __shared__ unsigned short Pl[4][16 * 40];   // wave-private P transpose
  const int t = threadIdx.x, lane = t & 31, wv = t >> 5;
  const int qt = blockIdx.x & 31;
  const int h  = (blockIdx.x >> 5) & 3;
  const int b  = blockIdx.x >> 7;
  const int ln = lane & 15, hf = lane >> 4;
  const int hcol = h * HD;
  const int qrow = b*SS + qt*64 + wv*16 + ln;

  const unsigned short* qp = &Q[qrow*256 + hcol + hf*8];
  v16bf qf0 = frag16(qp,      qp + 16);
  v16bf qf1 = frag16(qp + 32, qp + 48);

  auto stage = [&](int buf, int kt) {
    int row = t >> 2, cg = (t & 3) * 16;
    int g = (b*SS + kt + row)*256 + hcol + cg;
    cp16(&Kt[buf][row*72 + cg],     &K[g]);
    cp16(&Kt[buf][row*72 + cg + 8], &K[g + 8]);
    us8 v0 = *(const us8*)&V[g];
    us8 v1 = *(const us8*)&V[g + 8];
    #pragma unroll
    for (int i = 0; i < 8; ++i) Vt[buf][(cg + i)*40 + row]     = v0[i];
    #pragma unroll
    for (int i = 0; i < 8; ++i) Vt[buf][(cg + 8 + i)*40 + row] = v1[i];
  };

  float mx[8], l[8];
  v8f oacc[4] = {};
  #pragma unroll
  for (int r = 0; r < 8; ++r) { mx[r] = -1e30f; l[r] = 0.f; }

  stage(0, 0);
  cp_wait();
  __syncthreads();
  for (int ti = 0; ti < SS/32; ++ti) {
    const int cur = ti & 1;
    if (ti < SS/32 - 1) stage(1 - cur, (ti + 1) * 32);

    v16bf bk[2][2];
    #pragma unroll
    for (int j2 = 0; j2 < 2; ++j2) {
      const unsigned short* b0 = &Kt[cur][(j2*16 + ln)*72 + hf*8];
      bk[j2][0] = frag16(b0,      b0 + 16);
      bk[j2][1] = frag16(b0 + 32, b0 + 48);
    }
    v8f sacc[2] = {};
    #pragma unroll
    for (int j2 = 0; j2 < 2; ++j2) {
      sacc[j2] = __builtin_amdgcn_wmma_f32_16x16x32_bf16(false, qf0, false, bk[j2][0],
                                                         (short)0, sacc[j2], false, false);
      sacc[j2] = __builtin_amdgcn_wmma_f32_16x16x32_bf16(false, qf1, false, bk[j2][1],
                                                         (short)0, sacc[j2], false, false);
    }
    #pragma unroll
    for (int r = 0; r < 8; ++r) {
      float s0 = sacc[0][r] * 0.125f, s1 = sacc[1][r] * 0.125f;  // 1/sqrt(64)
      float tm = fmaxf(s0, s1);
      for (int msk = 1; msk < 16; msk <<= 1) tm = fmaxf(tm, __shfl_xor(tm, msk));
      float nm = fmaxf(mx[r], tm);
      float corr = __expf(mx[r] - nm);
      float p0 = __expf(s0 - nm), p1 = __expf(s1 - nm);
      float rs = p0 + p1;
      for (int msk = 1; msk < 16; msk <<= 1) rs += __shfl_xor(rs, msk);
      l[r] = l[r]*corr + rs;
      mx[r] = nm;
      #pragma unroll
      for (int j = 0; j < 4; ++j) oacc[j][r] *= corr;
      int m = r + hf*8;
      Pl[wv][m*40 + ln]      = f2bf(p0);
      Pl[wv][m*40 + 16 + ln] = f2bf(p1);
    }
    asm volatile("s_wait_dscnt 0x0" ::: "memory");   // wave-private LDS round trip
    const unsigned short* pp = &Pl[wv][ln*40 + hf*8];
    v16bf pa = frag16(pp, pp + 16);
    v16bf vb[4];
    #pragma unroll
    for (int j = 0; j < 4; ++j) {
      const unsigned short* vp = &Vt[cur][(j*16 + ln)*40 + hf*8];
      vb[j] = frag16(vp, vp + 16);
    }
    #pragma unroll
    for (int j = 0; j < 4; ++j)
      oacc[j] = __builtin_amdgcn_wmma_f32_16x16x32_bf16(false, pa, false, vb[j],
                                                        (short)0, oacc[j], false, false);
    cp_wait();
    __syncthreads();
  }
  #pragma unroll
  for (int j = 0; j < 4; ++j)
    #pragma unroll
    for (int r = 0; r < 8; ++r) {
      int row = b*SS + qt*64 + wv*16 + r + hf*8;
      O[row*256 + hcol + j*16 + ln] = f2bf(oacc[j][r] / l[r]);
    }
}

// ---------------------------------------------------------------- launch
extern "C" void kernel_launch(void* const* d_in, const int* in_sizes, int n_in,
                              void* d_out, int out_size, void* d_ws, size_t ws_size,
                              hipStream_t stream) {
  (void)in_sizes; (void)n_in; (void)out_size; (void)ws_size;
  const float* x    = (const float*)d_in[0];
  const float* cond = (const float*)d_in[1];
  const float* Wq_w = (const float*)d_in[2];
  const float* Wq_b = (const float*)d_in[3];
  const float* Wv_w = (const float*)d_in[4];
  const float* Wv_b = (const float*)d_in[5];
  const float* Wk_w = (const float*)d_in[6];
  const float* Wk_b = (const float*)d_in[7];
  const float* Wo_w = (const float*)d_in[8];
  const float* Wo_b = (const float*)d_in[9];
  float* out = (float*)d_out;

  char* ws = (char*)d_ws;
  size_t off = 0;
  auto alloc = [&](size_t bytes) -> void* {
    void* p = ws + off; off += (bytes + 255) & ~(size_t)255; return p;
  };
  unsigned short* xb = (unsigned short*)alloc((size_t)MTOT*256*2);
  unsigned short* wq = (unsigned short*)alloc(256*256*2);
  unsigned short* wvw= (unsigned short*)alloc(256*256*2);
  unsigned short* wk = (unsigned short*)alloc((size_t)32*256*256*2);
  unsigned short* wo = (unsigned short*)alloc(256*256*2);
  unsigned short* qb = (unsigned short*)alloc((size_t)MTOT*256*2);
  unsigned short* kb = (unsigned short*)alloc((size_t)MTOT*256*2);
  unsigned short* vb = (unsigned short*)alloc((size_t)MTOT*256*2);
  unsigned short* ob = (unsigned short*)alloc((size_t)MTOT*256*2);

  auto conv = [&](const float* s, unsigned short* d, int n) {
    int n4 = n / 4;
    k_convert<<<(n4 + 255)/256, 256, 0, stream>>>(s, d, n4);
  };
  conv(x,    xb, MTOT*256);
  conv(Wq_w, wq, 256*256);
  conv(Wv_w, wvw,256*256);
  conv(Wk_w, wk, 32*256*256);
  conv(Wo_w, wo, 256*256);

  dim3 ggrid(MTOT/64, 2);
  k_gemm<false><<<ggrid, 256, 0, stream>>>(xb, wq,  Wq_b, (void*)qb);
  k_gemm<false><<<ggrid, 256, 0, stream>>>(xb, wvw, Wv_b, (void*)vb);
  k_classkey  <<<ggrid, 256, 0, stream>>>(xb, wk, Wk_b, cond, kb);
  k_attn      <<<BB*NH*(SS/64), 128, 0, stream>>>(qb, kb, vb, ob);
  k_gemm<true><<<ggrid, 256, 0, stream>>>(ob, wo, Wo_b, (void*)out);
}
